// MambaDCLayer_9740985827497
// MI455X (gfx1250) — compile-verified, matrix operand order checked by reading, not run on hardware
//
#include <hip/hip_runtime.h>

// ---------------------------------------------------------------------------
// Mamba + depthwise-conv block for MI455X (gfx1250), wave32 / WMMA bf16.
//   B=8, L=4096, D=512, DI=1024, DS=16, R=32, KCONV=4, KC=31, M=B*L=32768
// GEMM staging: A tiles via Tensor Data Mover (TENSORcnt), B tiles via
// global_load_async_to_lds_b128 (ASYNCcnt), double-buffered LDS.
// ---------------------------------------------------------------------------

#define D_MODEL 512
#define D_INNER 1024
#define D_STATE 16
#define DT_RANK 32
#define SEQ_L   4096
#define NBATCH  8
#define MTOT    (NBATCH * SEQ_L)   // 32768 rows

typedef __attribute__((ext_vector_type(8)))  __bf16   v8bf;
typedef __attribute__((ext_vector_type(16))) __bf16   v16bf;
typedef __attribute__((ext_vector_type(8)))  float    v8f;
typedef __attribute__((ext_vector_type(4)))  unsigned v4u;
typedef __attribute__((ext_vector_type(8)))  int      v8i;
typedef __attribute__((ext_vector_type(4)))  int      v4i;

static __device__ __forceinline__ __bf16 to_bf16(float f) { return (__bf16)f; }

static __device__ __forceinline__ v16bf mk_frag(const __bf16* p) {
  // p points at the first 8-half K-chunk; the second chunk is +16 halves.
  v8bf lo = *(const v8bf*)p;
  v8bf hi = *(const v8bf*)(p + 16);
  v16bf r;
#pragma unroll
  for (int i = 0; i < 8; ++i) { r[i] = lo[i]; r[i + 8] = hi[i]; }
  return r;
}

// ---- CDNA5 async copy: global -> LDS, 16B per lane, tracked by ASYNCcnt ----
static __device__ __forceinline__ void async_b128(unsigned lds_addr,
                                                  const void* gaddr) {
  asm volatile("global_load_async_to_lds_b128 %0, %1, off"
               :: "v"(lds_addr), "v"((unsigned long long)(size_t)gaddr)
               : "memory");
}
static __device__ __forceinline__ void wait_async0() {
  asm volatile("s_wait_asynccnt 0x0" ::: "memory");
}

// ---- CDNA5 Tensor Data Mover: 2D bf16 tile (rows x 32), LDS-padded -------
// D# pad config: pad_interval code 3 (16 DWORDs = one 32-half row),
// pad_amount code 3 (4 DWORDs = 8 halves) -> LDS row stride 40 halves.
static __device__ __forceinline__ void tdm_load_tile(unsigned lds_addr,
                                                     const __bf16* gptr,
                                                     int K, int rows) {
  unsigned long long ga = (unsigned long long)(size_t)gptr;
  v4u g0 = {1u,                      // count=1, user descriptor
            lds_addr,                // lds_addr [63:32]
            (unsigned)(ga & 0xFFFFFFFFu),                       // global [95:64]
            (unsigned)(((ga >> 32) & 0x01FFFFFFu) | (2u << 30))}; // type=2
  unsigned td0 = (unsigned)K;        // tensor_dim0 (elements)
  unsigned td1 = (unsigned)rows;     // tensor_dim1
  unsigned long long s0 = (unsigned long long)K;  // dim0 stride (elements)
  v8i g1;
  g1[0] = (int)((1u << 16) |         // data_size = 2 bytes
                (1u << 20) |         // pad_enable
                (3u << 22) |         // pad_interval: 16 DWORDs
                (3u << 25));         // pad_amount: 4 DWORDs
  g1[1] = (int)((td0 & 0xFFFFu) << 16);
  g1[2] = (int)((td0 >> 16) | ((td1 & 0xFFFFu) << 16));
  g1[3] = (int)((td1 >> 16) | (32u << 16));       // tile_dim0 = 32
  g1[4] = (int)td1;                               // tile_dim1 = rows
  g1[5] = (int)(unsigned)(s0 & 0xFFFFFFFFu);
  g1[6] = (int)(unsigned)((s0 >> 32) & 0xFFFFu);
  g1[7] = 0;
  v4i z4 = {0, 0, 0, 0};
#if defined(__clang_major__) && (__clang_major__ >= 23)
  v8i z8 = {0, 0, 0, 0, 0, 0, 0, 0};
  __builtin_amdgcn_tensor_load_to_lds(g0, g1, z4, z4, z8, 0);
#else
  __builtin_amdgcn_tensor_load_to_lds(g0, g1, z4, z4, 0);
#endif
}

// ---------------------------------------------------------------------------
// fp32 -> bf16 weight conversion
// ---------------------------------------------------------------------------
__global__ __launch_bounds__(256) void k_cvt(const float* __restrict__ a,
                                             __bf16* __restrict__ o, int n) {
  int id = blockIdx.x * 256 + threadIdx.x;
  if (id < n) o[id] = to_bf16(a[id]);
}

// ---------------------------------------------------------------------------
// LayerNorm over D=512: one wave (32 lanes x 16 elems) per row, shfl reduce.
// ---------------------------------------------------------------------------
template <typename OutT>
__global__ __launch_bounds__(256) void k_ln(const float* __restrict__ x,
                                            const float* __restrict__ w,
                                            const float* __restrict__ b,
                                            OutT* __restrict__ o) {
  int wave = threadIdx.x >> 5, lane = threadIdx.x & 31;
  long row = (long)blockIdx.x * 8 + wave;
  const float* xr = x + row * D_MODEL;
  float v[16];
  float s = 0.f, s2 = 0.f;
#pragma unroll
  for (int i = 0; i < 16; ++i) {
    float t = xr[lane + 32 * i];
    v[i] = t; s += t; s2 += t * t;
  }
#pragma unroll
  for (int off = 16; off >= 1; off >>= 1) {
    s  += __shfl_xor(s,  off);
    s2 += __shfl_xor(s2, off);
  }
  float mu   = s * (1.f / D_MODEL);
  float var  = s2 * (1.f / D_MODEL) - mu * mu;
  float rstd = rsqrtf(var + 1e-5f);
  OutT* orow = o + row * D_MODEL;
#pragma unroll
  for (int i = 0; i < 16; ++i) {
    int c = lane + 32 * i;
    orow[c] = (OutT)((v[i] - mu) * rstd * w[c] + b[c]);
  }
}

// ---------------------------------------------------------------------------
// Tiled NT GEMM, bf16 in / f32 accumulate via v_wmma_f32_16x16x32_bf16.
//   C[M,N] = A[M,K] * Bw[N,K]^T  (both operands K-major => identical frag path)
// A tile staged by TDM (one descriptor per wave), B tile by async b128 copies,
// both double-buffered and overlapped with WMMA on the current buffer.
// EPI: 0 = plain store, 1 = x_proj (f32 + bf16 aux for n<DT_RANK),
//      2 = softplus(acc + bias[n]), 3 = residual add.
// ---------------------------------------------------------------------------
template <int BM, int BN, int WM, int WN, int EPI>
__global__ __launch_bounds__(256) void k_gemm(const __bf16* __restrict__ A,
                                              const __bf16* __restrict__ Bw,
                                              float* __restrict__ C,
                                              const float* __restrict__ resid,
                                              const float* __restrict__ bias,
                                              __bf16* __restrict__ aux,
                                              int N, int K) {
  constexpr int BK  = 32;
  constexpr int WTM = BM / WM / 16;
  constexpr int WTN = BN / WN / 16;
  constexpr int LDA = BK + 8;  // 40 halves = 80B rows: 16B aligned, bank-spread
  constexpr int BMW = BM / 8;  // A rows staged per wave via TDM
  __shared__ __align__(16) __bf16 As[2][BM * LDA];
  __shared__ __align__(16) __bf16 Bs[2][BN * LDA];

  const int t    = threadIdx.x;
  const int wave = t >> 5, lane = t & 31;
  const int wm   = wave % WM, wn = wave / WM;
  const int lr   = lane & 15;        // N (B,C,D) / M (A) index within tile
  const int lh   = lane >> 4;        // lane half selects K sub-chunk
  const long m0  = (long)blockIdx.y * BM;
  const long n0  = (long)blockIdx.x * BN;

  auto stage = [&](int sbuf, int k0) {
    // A: one TDM 2D descriptor per wave (BMW x 32 elements, padded rows)
    unsigned lds_a = (unsigned)(size_t)(&As[sbuf][0]) +
                     (unsigned)(wave * BMW * LDA * 2);
    tdm_load_tile(lds_a, &A[(m0 + wave * BMW) * (long)K + k0], K, BMW);
    // B: per-lane async 16B copies
    constexpr int CHB = BN * 4 / 256;
#pragma unroll
    for (int i = 0; i < CHB; ++i) {
      int c = t + i * 256;
      int r = c >> 2, k8 = (c & 3) << 3;
      unsigned lds_b = (unsigned)(size_t)(&Bs[sbuf][r * LDA + k8]);
      async_b128(lds_b, &Bw[(n0 + r) * (long)K + k0 + k8]);
    }
  };

  v8f zero = {0.f, 0.f, 0.f, 0.f, 0.f, 0.f, 0.f, 0.f};
  v8f acc[WTM][WTN];
#pragma unroll
  for (int i = 0; i < WTM; ++i)
#pragma unroll
    for (int j = 0; j < WTN; ++j) acc[i][j] = zero;

  stage(0, 0);
  __builtin_amdgcn_s_wait_tensorcnt(0);
  wait_async0();
  __syncthreads();

  int buf = 0;
  for (int k0 = 0; k0 < K; k0 += BK, buf ^= 1) {
    const bool more = (k0 + BK) < K;
    if (more) stage(buf ^ 1, k0 + BK);   // overlap DMA with WMMA below

    v16bf af[WTM], bfr[WTN];
#pragma unroll
    for (int i = 0; i < WTM; ++i) {
      int row = (wm * WTM + i) * 16 + lr;                 // M row
      af[i] = mk_frag(&As[buf][row * LDA + (lh << 3)]);
    }
#pragma unroll
    for (int j = 0; j < WTN; ++j) {
      int row = (wn * WTN + j) * 16 + lr;                 // N column (Bw row)
      bfr[j] = mk_frag(&Bs[buf][row * LDA + (lh << 3)]);
    }
#pragma unroll
    for (int i = 0; i < WTM; ++i)
#pragma unroll
      for (int j = 0; j < WTN; ++j)
        acc[i][j] = __builtin_amdgcn_wmma_f32_16x16x32_bf16(
            false, af[i], false, bfr[j], (short)0, acc[i][j], false, false);

    if (more) {
      __builtin_amdgcn_s_wait_tensorcnt(0);
      wait_async0();
      __syncthreads();
    }
  }

  // ---- epilogue: VGPR r of 16x16 tile holds M = r + 8*lh, N = lr ----
#pragma unroll
  for (int i = 0; i < WTM; ++i) {
#pragma unroll
    for (int j = 0; j < WTN; ++j) {
      const long mt = m0 + (long)(wm * WTM + i) * 16 + (lh << 3);
      const long nn = n0 + (long)(wn * WTN + j) * 16 + lr;
#pragma unroll
      for (int r = 0; r < 8; ++r) {
        long m = mt + r;
        float v = acc[i][j][r];
        if constexpr (EPI == 0) {
          C[m * (long)N + nn] = v;
        } else if constexpr (EPI == 1) {
          C[m * (long)N + nn] = v;
          if (nn < DT_RANK) aux[m * DT_RANK + nn] = to_bf16(v);
        } else if constexpr (EPI == 2) {
          float s = v + bias[nn];
          C[m * (long)N + nn] = (s > 20.f) ? s : log1pf(__expf(s));
        } else {
          C[m * (long)N + nn] = resid[m * (long)N + nn] + v;
        }
      }
    }
  }
}

// ---------------------------------------------------------------------------
// Causal depthwise conv (K=4) + bias + SiLU on xi = xz[:, :DI]; dual store.
// ---------------------------------------------------------------------------
__global__ __launch_bounds__(256) void k_conv4(const float* __restrict__ xz,
                                               const float* __restrict__ cw,
                                               const float* __restrict__ cb,
                                               float* __restrict__ xcf,
                                               __bf16* __restrict__ xcb) {
  long id = (long)blockIdx.x * 256 + threadIdx.x;   // M * DI threads
  long m  = id >> 10;
  int  d  = (int)(id & (D_INNER - 1));
  int  l  = (int)(m & (SEQ_L - 1));
  float acc = cb[d];
#pragma unroll
  for (int j = 0; j < 4; ++j) {
    int lj = l - 3 + j;
    if (lj >= 0) acc += cw[d * 4 + j] * xz[(m - 3 + j) * (2 * D_INNER) + d];
  }
  float sil = acc / (1.f + __expf(-acc));
  xcf[id] = sil;
  xcb[id] = to_bf16(sil);
}

// ---------------------------------------------------------------------------
// Selective scan: 16 lanes per (b,d) channel, one state per lane.
// Fused epilogue: (y + x*D) * silu(z) -> bf16 for out_proj.
// ---------------------------------------------------------------------------
__global__ __launch_bounds__(256) void k_scan(const float* __restrict__ xz,
                                              const float* __restrict__ xc,
                                              const float* __restrict__ dt,
                                              const float* __restrict__ xdb,
                                              const float* __restrict__ A_log,
                                              const float* __restrict__ Dp,
                                              __bf16* __restrict__ y) {
  int t = threadIdx.x;
  int g = blockIdx.x * 16 + (t >> 4);  // channel group id: b*DI + d
  int s = t & 15;                      // state index
  int b = g >> 10;
  int d = g & (D_INNER - 1);
  float Av = -__expf(A_log[d * D_STATE + s]);
  float Dv = Dp[d];
  float h  = 0.f;
  long mb  = (long)b * SEQ_L;
  for (int l = 0; l < SEQ_L; ++l) {
    long m    = mb + l;
    float dtv = dt[m * D_INNER + d];
    float xv  = xc[m * D_INNER + d];
    float Bv  = xdb[m * 64 + DT_RANK + s];
    float Cv  = xdb[m * 64 + DT_RANK + D_STATE + s];
    h = __expf(dtv * Av) * h + (dtv * xv) * Bv;
    float yv = h * Cv;
    yv += __shfl_xor(yv, 1);
    yv += __shfl_xor(yv, 2);
    yv += __shfl_xor(yv, 4);
    yv += __shfl_xor(yv, 8);
    if (s == 0) {
      float zv  = xz[m * (2 * D_INNER) + D_INNER + d];
      float sig = 1.f / (1.f + __expf(-zv));
      y[m * D_INNER + d] = to_bf16((yv + xv * Dv) * (zv * sig));
    }
  }
}

// ---------------------------------------------------------------------------
// 'same' depthwise conv (K=31) over time + bias + residual -> d_out.
// 64(L) x 64(C) output tile; 94 x 64 halo tile in LDS; 46-reg window/thread.
// ---------------------------------------------------------------------------
__global__ __launch_bounds__(256) void k_dconv(const float* __restrict__ h2,
                                               const float* __restrict__ attn,
                                               const float* __restrict__ w,
                                               const float* __restrict__ bias,
                                               float* __restrict__ out) {
  __shared__ float tile[94 * 64];
  int b  = blockIdx.z;
  int c0 = blockIdx.y * 64;
  int l0 = blockIdx.x * 64;
  int t  = threadIdx.x;
  for (int i = t; i < 94 * 64; i += 256) {
    int rr = i >> 6, cc = i & 63;
    int l = l0 - 15 + rr;
    float v = 0.f;
    if (l >= 0 && l < SEQ_L) v = h2[((long)b * SEQ_L + l) * D_MODEL + c0 + cc];
    tile[rr * 64 + cc] = v;
  }
  __syncthreads();
  int ci  = t & 63;
  int li0 = (t >> 6) * 16;
  float wv[31];
#pragma unroll
  for (int j = 0; j < 31; ++j) wv[j] = w[(c0 + ci) * 31 + j];
  float xv[46];
#pragma unroll
  for (int i = 0; i < 46; ++i) xv[i] = tile[(li0 + i) * 64 + ci];
  float bb = bias[c0 + ci];
#pragma unroll
  for (int k = 0; k < 16; ++k) {
    float acc = bb;
#pragma unroll
    for (int j = 0; j < 31; ++j) acc += wv[j] * xv[k + j];
    long m = (long)b * SEQ_L + l0 + li0 + k;
    out[m * D_MODEL + c0 + ci] = attn[m * D_MODEL + c0 + ci] + acc;
  }
}

// ---------------------------------------------------------------------------
// Workspace layout (bytes); attn/h2 reuse the dead xz region post-scan.
// ---------------------------------------------------------------------------
static constexpr size_t OFF_XN   = 0;                                  // 32MB bf16
static constexpr size_t OFF_WIN  = OFF_XN  + (size_t)MTOT * 512 * 2;   //  2MB
static constexpr size_t OFF_WOUT = OFF_WIN + 2048 * 512 * 2;           //  1MB
static constexpr size_t OFF_WXP  = OFF_WOUT + 512 * 1024 * 2;          // 128KB
static constexpr size_t OFF_WDT  = OFF_WXP + 64 * 1024 * 2;            //  64KB
static constexpr size_t OFF_DTR  = OFF_WDT + 1024 * 32 * 2;            //  2MB bf16
static constexpr size_t OFF_XDB  = OFF_DTR + (size_t)MTOT * 32 * 2;    //  8MB f32
static constexpr size_t OFF_XZ   = OFF_XDB + (size_t)MTOT * 64 * 4;    // 256MB f32
static constexpr size_t OFF_XCF  = OFF_XZ  + (size_t)MTOT * 2048 * 4;  // 128MB f32
static constexpr size_t OFF_XCB  = OFF_XCF + (size_t)MTOT * 1024 * 4;  //  64MB bf16
static constexpr size_t OFF_DT   = OFF_XCB + (size_t)MTOT * 1024 * 2;  // 128MB f32
static constexpr size_t OFF_Y    = OFF_DT  + (size_t)MTOT * 1024 * 4;  //  64MB bf16
static constexpr size_t OFF_ATTN = OFF_XZ;                             // reuse
static constexpr size_t OFF_H2   = OFF_XZ + (size_t)MTOT * 512 * 4;    // reuse

extern "C" void kernel_launch(void* const* d_in, const int* in_sizes, int n_in,
                              void* d_out, int out_size, void* d_ws,
                              size_t ws_size, hipStream_t stream) {
  (void)in_sizes; (void)n_in; (void)out_size; (void)ws_size;
  const float* x         = (const float*)d_in[0];
  const float* ln1_w     = (const float*)d_in[1];
  const float* ln1_b     = (const float*)d_in[2];
  const float* in_proj_w = (const float*)d_in[3];
  const float* conv_w    = (const float*)d_in[4];
  const float* conv_b    = (const float*)d_in[5];
  const float* x_proj_w  = (const float*)d_in[6];
  const float* dt_proj_w = (const float*)d_in[7];
  const float* dt_proj_b = (const float*)d_in[8];
  const float* A_log     = (const float*)d_in[9];
  const float* Dp        = (const float*)d_in[10];
  const float* out_projw = (const float*)d_in[11];
  const float* ln2_w     = (const float*)d_in[12];
  const float* ln2_b     = (const float*)d_in[13];
  const float* dc_w      = (const float*)d_in[14];
  const float* dc_b      = (const float*)d_in[15];

  char* ws = (char*)d_ws;
  __bf16* xn   = (__bf16*)(ws + OFF_XN);
  __bf16* w_in = (__bf16*)(ws + OFF_WIN);
  __bf16* w_out= (__bf16*)(ws + OFF_WOUT);
  __bf16* w_xp = (__bf16*)(ws + OFF_WXP);
  __bf16* w_dt = (__bf16*)(ws + OFF_WDT);
  __bf16* dtr  = (__bf16*)(ws + OFF_DTR);
  float*  xdb  = (float*)(ws + OFF_XDB);
  float*  xz   = (float*)(ws + OFF_XZ);
  float*  xcf  = (float*)(ws + OFF_XCF);
  __bf16* xcb  = (__bf16*)(ws + OFF_XCB);
  float*  dtb  = (float*)(ws + OFF_DT);
  __bf16* ybf  = (__bf16*)(ws + OFF_Y);
  float*  attn = (float*)(ws + OFF_ATTN);
  float*  h2   = (float*)(ws + OFF_H2);
  float*  out  = (float*)d_out;

  // weight conversions (tiny)
  k_cvt<<<4096, 256, 0, stream>>>(in_proj_w, w_in, 2048 * 512);
  k_cvt<<<256,  256, 0, stream>>>(x_proj_w,  w_xp, 64 * 1024);
  k_cvt<<<128,  256, 0, stream>>>(dt_proj_w, w_dt, 1024 * 32);
  k_cvt<<<2048, 256, 0, stream>>>(out_projw, w_out, 512 * 1024);

  // LN1 -> bf16
  k_ln<__bf16><<<MTOT / 8, 256, 0, stream>>>(x, ln1_w, ln1_b, xn);

  // in_proj: [M,512] x [2048,512]^T -> xz [M,2048]
  k_gemm<128, 128, 2, 4, 0><<<dim3(2048 / 128, MTOT / 128), 256, 0, stream>>>(
      xn, w_in, xz, nullptr, nullptr, nullptr, 2048, 512);

  // conv4 + SiLU on xi
  k_conv4<<<(MTOT * D_INNER) / 256, 256, 0, stream>>>(xz, conv_w, conv_b, xcf, xcb);

  // x_proj: [M,1024] x [64,1024]^T -> xdb [M,64]; bf16 aux of dt-rank slice
  k_gemm<128, 64, 4, 2, 1><<<dim3(1, MTOT / 128), 256, 0, stream>>>(
      xcb, w_xp, xdb, nullptr, nullptr, dtr, 64, 1024);

  // dt_proj + softplus: [M,32] x [1024,32]^T -> dt [M,1024]
  k_gemm<128, 128, 2, 4, 2><<<dim3(1024 / 128, MTOT / 128), 256, 0, stream>>>(
      dtr, w_dt, dtb, nullptr, dt_proj_b, nullptr, 1024, 32);

  // selective scan -> y*silu(z) as bf16
  k_scan<<<(NBATCH * D_INNER) / 16, 256, 0, stream>>>(xz, xcf, dtb, xdb,
                                                      A_log, Dp, ybf);

  // out_proj + residual: [M,1024] x [512,1024]^T + x -> attn [M,512]
  k_gemm<128, 128, 2, 4, 3><<<dim3(512 / 128, MTOT / 128), 256, 0, stream>>>(
      ybf, w_out, attn, x, nullptr, nullptr, 512, 1024);

  // LN2 -> f32
  k_ln<float><<<MTOT / 8, 256, 0, stream>>>(attn, ln2_w, ln2_b, h2);

  // depthwise conv31 + bias + residual -> d_out
  k_dconv<<<dim3(SEQ_L / 64, D_MODEL / 64, NBATCH), 256, 0, stream>>>(
      h2, attn, dc_w, dc_b, out);
}
